// BPECTC_32530082300566
// MI455X (gfx1250) — matrix operand order repaired and checked by reading, not compiled, not run on hardware
//
#include <hip/hip_runtime.h>

// Problem constants (fixed by the reference harness)
#define BQ 16
#define TT 800
#define DD 512
#define VV 5000
#define LL 100
#define BT (BQ * TT)      // 12800 rows
#define SS (2 * LL + 1)   // 201 extended states
#define NE 208            // padded ext columns (13 * 16)
#define NEG_INF -1e30f

typedef __bf16 v16bf __attribute__((ext_vector_type(16)));
typedef __bf16 v8bf  __attribute__((ext_vector_type(8)));
typedef float  v8f   __attribute__((ext_vector_type(8)));

__device__ __forceinline__ int imin(int a, int b) { return a < b ? a : b; }

// ---------------------------------------------------------------------------
// 1) Split hs_pad fp32 -> bf16 hi + lo residual
// ---------------------------------------------------------------------------
__global__ void k_split_hs(const float* __restrict__ in,
                           __bf16* __restrict__ hi, __bf16* __restrict__ lo) {
  size_t i = (size_t)blockIdx.x * blockDim.x + threadIdx.x;
  if (i >= (size_t)BT * DD) return;
  float x = in[i];
  __bf16 h = (__bf16)x;
  hi[i] = h;
  lo[i] = (__bf16)(x - (float)h);
}

// ---------------------------------------------------------------------------
// 2) Transpose W [D,V] -> Wt [V,D] with bf16 hi/lo split (LDS tiled)
// ---------------------------------------------------------------------------
__global__ void k_transW(const float* __restrict__ W,
                         __bf16* __restrict__ Wt_hi, __bf16* __restrict__ Wt_lo) {
  __shared__ float tile[16][17];
  int tx = threadIdx.x, ty = threadIdx.y;
  int v0 = blockIdx.x * 16, d0 = blockIdx.y * 16;
  int v = v0 + tx;
  int d = d0 + ty;
  if (v < VV) tile[ty][tx] = W[(size_t)d * VV + v];
  __syncthreads();
  int vv = v0 + ty, dd = d0 + tx;
  if (vv < VV) {
    float x = tile[tx][ty];
    __bf16 h = (__bf16)x;
    Wt_hi[(size_t)vv * DD + dd] = h;
    Wt_lo[(size_t)vv * DD + dd] = (__bf16)(x - (float)h);
  }
}

// ---------------------------------------------------------------------------
// 3) Gather per-utterance extended-label columns of Wt into We [B, NE, D]
//    ext[s] = 0 (blank) for even s, labels[(s-1)/2] for odd s; zero pad s>=S
// ---------------------------------------------------------------------------
__global__ void k_gather_ext(const __bf16* __restrict__ Wt_hi,
                             const __bf16* __restrict__ Wt_lo,
                             const int* __restrict__ labels,
                             __bf16* __restrict__ We_hi, __bf16* __restrict__ We_lo) {
  int b = blockIdx.x, s = blockIdx.y;
  int tid = threadIdx.x;  // 128 threads
  size_t dst = ((size_t)b * NE + s) * DD;
  if (s >= SS) {
    for (int i = tid; i < DD; i += 128) {
      We_hi[dst + i] = (__bf16)0.f;
      We_lo[dst + i] = (__bf16)0.f;
    }
    return;
  }
  int col = (s & 1) ? labels[b * LL + ((s - 1) >> 1)] : 0;
  size_t src = (size_t)col * DD;
  for (int i = tid; i < DD; i += 128) {
    We_hi[dst + i] = Wt_hi[src + i];
    We_lo[dst + i] = Wt_lo[src + i];
  }
}

// ---------------------------------------------------------------------------
// WMMA helper: one 16x16 fp32 tile accumulated over K=512 with bf16x2 split
// (3 WMMAs per 32-wide K slice: ah*bh + ah*bl + al*bh)
// ---------------------------------------------------------------------------
__device__ __forceinline__ v8f gemm_row_tile(const __bf16* __restrict__ ah_base,
                                             const __bf16* __restrict__ al_base,
                                             const __bf16* __restrict__ bh_base,
                                             const __bf16* __restrict__ bl_base,
                                             int asel, int bsel) {
  v8f c = {};
#pragma unroll 4
  for (int kt = 0; kt < 16; kt++) {
    const int k0 = kt * 32;
    union { v16bf v; v8bf h[2]; } ah, al;
    ah.h[0] = *(const v8bf*)(ah_base + k0 + asel);
    ah.h[1] = *(const v8bf*)(ah_base + k0 + asel + 8);
    al.h[0] = *(const v8bf*)(al_base + k0 + asel);
    al.h[1] = *(const v8bf*)(al_base + k0 + asel + 8);
    v16bf bh = *(const v16bf*)(bh_base + k0 + bsel);
    v16bf bl = *(const v16bf*)(bl_base + k0 + bsel);
    c = __builtin_amdgcn_wmma_f32_16x16x32_bf16(false, ah.v, false, bh, (short)0, c, false, false);
    c = __builtin_amdgcn_wmma_f32_16x16x32_bf16(false, ah.v, false, bl, (short)0, c, false, false);
    c = __builtin_amdgcn_wmma_f32_16x16x32_bf16(false, al.v, false, bh, (short)0, c, false, false);
  }
  return c;
}

// ---------------------------------------------------------------------------
// 4) Fused GEMM + online log-sum-exp over V. One block = 16 rows of (b,t).
//    4 waves stride the 313 N-tiles; never materializes logits.
// ---------------------------------------------------------------------------
__global__ __launch_bounds__(128) void k_gemm_lse(
    const __bf16* __restrict__ hs_hi, const __bf16* __restrict__ hs_lo,
    const __bf16* __restrict__ Wt_hi, const __bf16* __restrict__ Wt_lo,
    const float* __restrict__ bias, float* __restrict__ lse) {
  const int lane = threadIdx.x & 31;
  const int wave = threadIdx.x >> 5;
  const int mt = blockIdx.x;  // 0..799
  const size_t arow = (size_t)(mt * 16 + (lane & 15)) * DD;
  const int asel = (lane >> 4) * 16;
  const int bsel = (lane >> 4) * 16;

  float mrow[8], srow[8];
#pragma unroll
  for (int r = 0; r < 8; r++) { mrow[r] = NEG_INF; srow[r] = 0.f; }

  const int NT = (VV + 15) / 16;  // 313
  for (int j = wave; j < NT; j += 4) {
    const int nb = j * 16;
    const int n = imin(nb + (lane & 15), VV - 1);
    const bool colvalid = (nb + (lane & 15)) < VV;
    const size_t brow = (size_t)n * DD;
    const float bcol = bias[n];
    v8f c = gemm_row_tile(hs_hi + arow, hs_lo + arow, Wt_hi + brow, Wt_lo + brow,
                          asel, bsel);
    if (colvalid) {
#pragma unroll
      for (int r = 0; r < 8; r++) {
        float x = c[r] + bcol;
        float mn = fmaxf(mrow[r], x);
        srow[r] = srow[r] * __expf(mrow[r] - mn) + __expf(x - mn);
        mrow[r] = mn;
      }
    }
  }
  // reduce (m,s) across the 16 column-lanes inside each half-wave
#pragma unroll
  for (int r = 0; r < 8; r++) {
    for (int mask = 8; mask >= 1; mask >>= 1) {
      float mo = __shfl_xor(mrow[r], mask, 32);
      float so = __shfl_xor(srow[r], mask, 32);
      float mn = fmaxf(mrow[r], mo);
      srow[r] = srow[r] * __expf(mrow[r] - mn) + so * __expf(mo - mn);
      mrow[r] = mn;
    }
  }
  __shared__ float sm[4][16], ss[4][16];
  if ((lane & 15) == 0) {
    int rbase = (lane >> 4) * 8;
#pragma unroll
    for (int r = 0; r < 8; r++) { sm[wave][rbase + r] = mrow[r]; ss[wave][rbase + r] = srow[r]; }
  }
  __syncthreads();
  if (threadIdx.x < 16) {
    int row = threadIdx.x;
    float m = sm[0][row], s = ss[0][row];
    for (int w = 1; w < 4; w++) {
      float mo = sm[w][row], so = ss[w][row];
      float mn = fmaxf(m, mo);
      s = s * __expf(m - mn) + so * __expf(mo - mn);
      m = mn;
    }
    lse[mt * 16 + row] = m + __logf(s);
  }
}

// ---------------------------------------------------------------------------
// 5) Small GEMM for the 201 extended-label logits: [800,512] x [512,208] per b
// ---------------------------------------------------------------------------
__global__ __launch_bounds__(32) void k_gemm_ext(
    const __bf16* __restrict__ hs_hi, const __bf16* __restrict__ hs_lo,
    const __bf16* __restrict__ We_hi, const __bf16* __restrict__ We_lo,
    float* __restrict__ extlog) {
  const int b = blockIdx.x, mt = blockIdx.y, nt = blockIdx.z;
  const int lane = threadIdx.x;
  const size_t arow = ((size_t)b * TT + mt * 16 + (lane & 15)) * DD;
  const size_t brow = ((size_t)b * NE + nt * 16 + (lane & 15)) * DD;
  const int asel = (lane >> 4) * 16;
  const int bsel = (lane >> 4) * 16;
  v8f c = gemm_row_tile(hs_hi + arow, hs_lo + arow, We_hi + brow, We_lo + brow,
                        asel, bsel);
  const int col = nt * 16 + (lane & 15);
  const int rowbase = mt * 16 + (lane >> 4) * 8;
#pragma unroll
  for (int r = 0; r < 8; r++)
    extlog[((size_t)b * TT + rowbase + r) * NE + col] = c[r];
}

// ---------------------------------------------------------------------------
// 6) CTC forward DP: one wave32 per utterance, 7 states/lane (s = i*32+lane),
//    neighbor states via shuffles, no barriers in the 800-step loop.
// ---------------------------------------------------------------------------
__global__ __launch_bounds__(32) void k_ctc_dp(
    const float* __restrict__ extlog, const float* __restrict__ lse,
    const float* __restrict__ bias, const int* __restrict__ labels,
    const int* __restrict__ hlens, const int* __restrict__ label_lens,
    float* __restrict__ tot) {
  const int b = blockIdx.x;
  const int lane = threadIdx.x;
  const int NI = 7;  // 7*32 = 224 >= 201 states
  const int hlen = hlens[b];

  int extv[NI];
  bool skipv[NI];
  float bb[NI];
#pragma unroll
  for (int i = 0; i < NI; i++) {
    int s = i * 32 + lane;
    int e = 0;
    if (s < SS && (s & 1)) e = labels[b * LL + ((s - 1) >> 1)];
    extv[i] = e;
    bb[i] = (s < SS) ? bias[e] : 0.f;
    bool sk = false;
    if (s >= 2 && s < SS && (s & 1)) {
      int ep = labels[b * LL + ((s - 1) >> 1) - 1];
      sk = (extv[i] != 0) && (extv[i] != ep);
    }
    skipv[i] = sk;
  }

  const float* lprow = extlog + (size_t)b * TT * NE;
  float alpha[NI];
  {
    float lse0 = lse[b * TT];
#pragma unroll
    for (int i = 0; i < NI; i++) {
      int s = i * 32 + lane;
      float lp = (s < SS) ? lprow[s] + bb[i] - lse0 : NEG_INF;
      alpha[i] = (s < 2) ? lp : NEG_INF;
    }
  }

  for (int t = 1; t < TT; t++) {
    const float* lpt = lprow + (size_t)t * NE;
    const float lset = lse[b * TT + t];
    float p1[NI], p2[NI];
#pragma unroll
    for (int i = 0; i < NI; i++) {
      float up1 = __shfl_up(alpha[i], 1, 32);
      float w31 = (i > 0) ? __shfl(alpha[i - 1], 31, 32) : NEG_INF;
      p1[i] = (lane == 0) ? w31 : up1;
      float up2 = __shfl_up(alpha[i], 2, 32);
      float wlo = (i > 0) ? __shfl(alpha[i - 1], (30 + lane) & 31, 32) : NEG_INF;
      p2[i] = (lane < 2) ? wlo : up2;
    }
    const bool active = (t < hlen);
#pragma unroll
    for (int i = 0; i < NI; i++) {
      int s = i * 32 + lane;
      float lp = (s < SS) ? lpt[s] + bb[i] - lset : NEG_INF;
      float c2 = skipv[i] ? p2[i] : NEG_INF;
      float m = fmaxf(alpha[i], fmaxf(p1[i], c2));
      float sum = __expf(alpha[i] - m) + __expf(p1[i] - m) + __expf(c2 - m);
      float anew = m + __logf(sum) + lp;
      alpha[i] = active ? anew : alpha[i];
    }
  }

  __shared__ float alds[NI * 32];
#pragma unroll
  for (int i = 0; i < NI; i++) alds[i * 32 + lane] = alpha[i];
  __syncthreads();
  if (lane == 0) {
    int ll = label_lens[b];
    float a1 = alds[2 * ll], a0 = alds[2 * ll - 1];
    float m = fmaxf(a1, a0);
    tot[b] = m + __logf(__expf(a1 - m) + __expf(a0 - m));
  }
}

// ---------------------------------------------------------------------------
// 7) mean(-tot)
// ---------------------------------------------------------------------------
__global__ void k_finalize(const float* __restrict__ tot, float* __restrict__ out) {
  if (threadIdx.x == 0) {
    float s = 0.f;
    for (int b = 0; b < BQ; b++) s -= tot[b];
    out[0] = s / (float)BQ;
  }
}

// ---------------------------------------------------------------------------
extern "C" void kernel_launch(void* const* d_in, const int* in_sizes, int n_in,
                              void* d_out, int out_size, void* d_ws, size_t ws_size,
                              hipStream_t stream) {
  const float* hs = (const float*)d_in[0];
  const float* W = (const float*)d_in[1];
  const float* bias = (const float*)d_in[2];
  const int* hlens = (const int*)d_in[3];
  const int* labels = (const int*)d_in[4];
  const int* label_lens = (const int*)d_in[5];
  float* out = (float*)d_out;

  char* p = (char*)d_ws;
  auto take = [&](size_t bytes) -> char* {
    char* r = p;
    p += (bytes + 255) & ~(size_t)255;
    return r;
  };
  __bf16* hs_hi = (__bf16*)take((size_t)BT * DD * 2);
  __bf16* hs_lo = (__bf16*)take((size_t)BT * DD * 2);
  __bf16* Wt_hi = (__bf16*)take((size_t)VV * DD * 2);
  __bf16* Wt_lo = (__bf16*)take((size_t)VV * DD * 2);
  __bf16* We_hi = (__bf16*)take((size_t)BQ * NE * DD * 2);
  __bf16* We_lo = (__bf16*)take((size_t)BQ * NE * DD * 2);
  float* lse = (float*)take((size_t)BT * 4);
  float* extlog = (float*)take((size_t)BQ * TT * NE * 4);
  float* tot = (float*)take(64);

  k_split_hs<<<((size_t)BT * DD + 255) / 256, 256, 0, stream>>>(hs, hs_hi, hs_lo);
  k_transW<<<dim3((VV + 15) / 16, DD / 16), dim3(16, 16), 0, stream>>>(W, Wt_hi, Wt_lo);
  k_gather_ext<<<dim3(BQ, NE), 128, 0, stream>>>(Wt_hi, Wt_lo, labels, We_hi, We_lo);
  k_gemm_lse<<<BT / 16, 128, 0, stream>>>(hs_hi, hs_lo, Wt_hi, Wt_lo, bias, lse);
  k_gemm_ext<<<dim3(BQ, TT / 16, NE / 16), 32, 0, stream>>>(hs_hi, hs_lo, We_hi, We_lo, extlog);
  k_ctc_dp<<<BQ, 32, 0, stream>>>(extlog, lse, bias, labels, hlens, label_lens, tot);
  k_finalize<<<1, 32, 0, stream>>>(tot, out);
}
